// OuterProductMean_962072674751
// MI455X (gfx1250) — matrix operand order repaired and checked by reading, not compiled, not run on hardware
//
#include <hip/hip_runtime.h>

// ---------------------------------------------------------------------------
// OuterProductMean fused kernels for gfx1250 (MI455X), wave32 + WMMA bf16
// + TDM (tensor_load_to_lds) staging.
// Shapes: b=1, s=128, n_res=256, c_s=256, c=32, c_z=128.
// ---------------------------------------------------------------------------

typedef __attribute__((ext_vector_type(16))) __bf16 v16bf;
typedef __attribute__((ext_vector_type(8)))  float  v8f;
typedef unsigned int u32x4 __attribute__((ext_vector_type(4)));
typedef int          i32x8 __attribute__((ext_vector_type(8)));
typedef int          i32x4 __attribute__((ext_vector_type(4)));

#define S_DIM   128
#define NRES    256
#define CS      256
#define C_DIM   32
#define CZ      128
#define MTOT    8192          // n_res * c  (rows of A/B bf16, m-major layout [m][s])
#define LDAB    40            // padded LDS k-stride for sA/sB (32 + 4 DWORDs pad)
#define SMLD    1032          // padded LDS k-stride for sM

__device__ __forceinline__ unsigned short f2bf(float f) {
    unsigned int u = __builtin_bit_cast(unsigned int, f);
    unsigned int r = (u + 0x7FFFu + ((u >> 16) & 1u)) >> 16;  // RNE
    return (unsigned short)r;
}

// ---------------------------------------------------------------------------
// Kernel 1: LayerNorm + left/right projections -> bf16 A,B  m-major [i*32+c][s]
// One block (256 thr) per (s,i) row.
// ---------------------------------------------------------------------------
__global__ void opm_ln_proj_kernel(const float* __restrict__ msa,
                                   const float* __restrict__ gamma,
                                   const float* __restrict__ beta,
                                   const float* __restrict__ Wl,
                                   const float* __restrict__ bl,
                                   const float* __restrict__ Wr,
                                   const float* __restrict__ br,
                                   unsigned short* __restrict__ Abf,
                                   unsigned short* __restrict__ Bbf) {
    __shared__ float red[256];
    __shared__ float sxn[256];
    __shared__ float pr[4][64];

    const int row = blockIdx.x;        // row = s*256 + i
    const int t   = threadIdx.x;

    float x = msa[row * CS + t];

    // mean
    red[t] = x;
    __syncthreads();
    #pragma unroll
    for (int off = 128; off > 0; off >>= 1) {
        if (t < off) red[t] += red[t + off];
        __syncthreads();
    }
    const float mu = red[0] * (1.0f / 256.0f);
    __syncthreads();

    // variance
    const float v = x - mu;
    red[t] = v * v;
    __syncthreads();
    #pragma unroll
    for (int off = 128; off > 0; off >>= 1) {
        if (t < off) red[t] += red[t + off];
        __syncthreads();
    }
    const float rstd = rsqrtf(red[0] * (1.0f / 256.0f) + 1e-5f);

    sxn[t] = v * rstd * gamma[t] + beta[t];
    __syncthreads();

    // projections: 64 outputs (32 left + 32 right), 4 k-chunks of 64
    const int o    = t & 63;
    const int ch   = t >> 6;
    const int c    = o & 31;
    const int side = o >> 5;
    const float* W = side ? Wr : Wl;

    float p = 0.0f;
    const int kb = ch * 64;
    #pragma unroll 8
    for (int k = 0; k < 64; ++k)
        p += sxn[kb + k] * W[(kb + k) * C_DIM + c];
    pr[ch][o] = p;
    __syncthreads();

    if (ch == 0) {
        float tot = pr[0][o] + pr[1][o] + pr[2][o] + pr[3][o] + (side ? br[c] : bl[c]);
        unsigned short* dst = side ? Bbf : Abf;
        const int s = row >> 8;       // / 256
        const int i = row & 255;
        dst[(i * C_DIM + c) * S_DIM + s] = f2bf(tot);   // m-major
    }
}

// ---------------------------------------------------------------------------
// Kernel 1b: W_out f32 [1024][128] -> bf16 transposed [128][1024]
// ---------------------------------------------------------------------------
__global__ void opm_wcvt_kernel(const float* __restrict__ Wout,
                                unsigned short* __restrict__ WoutT) {
    const int idx = blockIdx.x * 256 + threadIdx.x;   // idx = k*128 + n (coalesced read)
    const int k = idx >> 7;
    const int n = idx & 127;
    WoutT[n * 1024 + k] = f2bf(Wout[idx]);
}

// ---------------------------------------------------------------------------
// Vectorized fragment builders: two 16B loads per lane.
// ---------------------------------------------------------------------------
__device__ __forceinline__ v16bf load_frag_pair(const unsigned short* p0,
                                                const unsigned short* p1) {
    union { v16bf v; uint4 q[2]; } r;
    r.q[0] = *(const uint4*)p0;
    r.q[1] = *(const uint4*)p1;
    return r.v;
}

// A-frag 16x32 from m-major source: elem(m,k) = base[m*ld + k]
__device__ __forceinline__ v16bf build_a(const unsigned short* base, int lane,
                                         int m0, int k0, int ld) {
    const int m  = m0 + (lane & 15);
    const int kb = k0 + (lane >> 4) * 8;
    return load_frag_pair(base + m * ld + kb, base + m * ld + kb + 16);
}

// B-frag 32x16 from n-major source: elem(k,n) = base[n*ld + k]
__device__ __forceinline__ v16bf build_b(const unsigned short* base, int lane,
                                         int n0, int k0, int ld) {
    const int n  = n0 + (lane & 15);
    const int kb = k0 + (lane >> 4) * 8;
    return load_frag_pair(base + n * ld + kb, base + n * ld + kb + 16);
}

// ---------------------------------------------------------------------------
// TDM: 2D tile load Global -> LDS.
// Tensor: [8192][128] bf16 (row stride 128 elems). Tile: 128 rows x 32 elems.
// pad_enable: after every 16 DWORDs (one 64B tile row) insert 4 DWORDs ->
// LDS row stride = 40 ushorts (= LDAB).
// ---------------------------------------------------------------------------
__device__ __forceinline__ void tdm_load_tile(unsigned lds_off,
                                              const unsigned short* gbase,
                                              unsigned long long elem_off) {
    const unsigned long long ga =
        (unsigned long long)(uintptr_t)gbase + elem_off * 2ull;
    u32x4 g0;
    g0[0] = 1u;                                            // count=1, user desc
    g0[1] = lds_off;                                       // lds_addr
    g0[2] = (unsigned)ga;                                  // global_addr[31:0]
    g0[3] = (unsigned)((ga >> 32) & 0x1FFFFFFull) | (2u << 30);  // [56:32] | type=2

    i32x8 g1;
    g1[0] = (int)((1u << 16) | (1u << 20) | (3u << 22) | (3u << 25));
            // data_size=2B, pad_enable=1, pad_interval=16DW, pad_amount=4DW
    g1[1] = (int)(128u << 16);        // tensor_dim0 = 128 (low16 @ bits63:48)
    g1[2] = (int)(8192u << 16);       // dim0 hi16=0 | tensor_dim1 low16 = 8192
    g1[3] = (int)(32u << 16);         // dim1 hi16=0 | tile_dim0 = 32
    g1[4] = (int)(128u);              // tile_dim1 = 128 | tile_dim2 = 0
    g1[5] = (int)(128u);              // tensor_dim0_stride = 128 (low32)
    g1[6] = 0;                        // stride0 hi16 | tensor_dim1_stride low16
    g1[7] = 0;

    const i32x4 z4 = {0, 0, 0, 0};                 // groups 2/3 unused (2D tile)
    const i32x8 z8 = {0, 0, 0, 0, 0, 0, 0, 0};     // trailing group (unused)
    __builtin_amdgcn_tensor_load_to_lds(g0, g1, z4, z4, z8, 0);
}

// ---------------------------------------------------------------------------
// Kernel 2: fused  mean = (A^T B)/s  (per 4x4 (i,j) tile)  ->  mean @ W_out
// Block: 256 threads = 8 waves.  Grid: 64 x 64.
//   GEMM1: M1=128 (4 i * 32 c), N1=128 (4 j * 32 d), K=128 (s)
//   GEMM2: M2=16 (pairs), K=1024 (c*d), N=128 (c_z), one n-tile per wave
// ---------------------------------------------------------------------------
__global__ void opm_fused_kernel(const unsigned short* __restrict__ Abf,
                                 const unsigned short* __restrict__ Bbf,
                                 const unsigned short* __restrict__ WoutT,
                                 const float* __restrict__ b_out,
                                 float* __restrict__ out) {
    __shared__ __align__(16) unsigned short sA[128 * LDAB];   // 10 KB, m-major
    __shared__ __align__(16) unsigned short sB[128 * LDAB];   // 10 KB, n-major
    __shared__ __align__(16) unsigned short sM[16 * SMLD];    // 33 KB, pair-major

    const int t    = threadIdx.x;
    const int lane = t & 31;
    const int wave = t >> 5;
    const int i0   = blockIdx.y * 4;
    const int j0   = blockIdx.x * 4;

    // warm W_out^T into L2
    if (t < CZ) __builtin_prefetch(WoutT + t * 1024, 0, 1);

    const v8f vzero = {0.f, 0.f, 0.f, 0.f, 0.f, 0.f, 0.f, 0.f};
    v8f acc[2][4];
    #pragma unroll
    for (int mi = 0; mi < 2; ++mi)
        #pragma unroll
        for (int ni = 0; ni < 4; ++ni) acc[mi][ni] = vzero;

    const int mrow0 = (wave & 3) * 32;   // 2 m-tiles per wave
    const int nc0   = (wave >> 2) * 64;  // 4 n-tiles per wave

    // ---------------- GEMM1: accumulate over 4 chunks of K=32 ----------------
    #pragma unroll 1
    for (int kc = 0; kc < 4; ++kc) {
        const int k0 = kc * 32;
        if (wave == 0) {
            tdm_load_tile((unsigned)(uintptr_t)sA, Abf,
                          (unsigned long long)(i0 * C_DIM) * S_DIM + k0);
            tdm_load_tile((unsigned)(uintptr_t)sB, Bbf,
                          (unsigned long long)(j0 * C_DIM) * S_DIM + k0);
            __builtin_amdgcn_s_wait_tensorcnt(0);
        }
        __syncthreads();

        v16bf af[2], bfr[4];
        #pragma unroll
        for (int mi = 0; mi < 2; ++mi)
            af[mi] = build_a(sA, lane, mrow0 + mi * 16, 0, LDAB);
        #pragma unroll
        for (int ni = 0; ni < 4; ++ni)
            bfr[ni] = build_b(sB, lane, nc0 + ni * 16, 0, LDAB);

        #pragma unroll
        for (int mi = 0; mi < 2; ++mi)
            #pragma unroll
            for (int ni = 0; ni < 4; ++ni)
                acc[mi][ni] = __builtin_amdgcn_wmma_f32_16x16x32_bf16(
                    false, af[mi], false, bfr[ni], (short)0, acc[mi][ni],
                    false, false);
        __syncthreads();
    }

    // ------- scale by 1/s, write mean tile to LDS as bf16 [pair][c*32+d] -----
    const float inv_s = 1.0f / (float)S_DIM;
    #pragma unroll
    for (int mi = 0; mi < 2; ++mi) {
        #pragma unroll
        for (int ni = 0; ni < 4; ++ni) {
            #pragma unroll
            for (int r = 0; r < 8; ++r) {
                const int mG = mrow0 + mi * 16 + (lane >> 4) * 8 + r;
                const int nG = nc0 + ni * 16 + (lane & 15);
                const int p  = (mG >> 5) * 4 + (nG >> 5);       // i_loc*4 + j_loc
                const int k  = (mG & 31) * 32 + (nG & 31);      // c*32 + d
                sM[p * SMLD + k] = f2bf(acc[mi][ni][r] * inv_s);
            }
        }
    }
    __syncthreads();

    // ---------------- GEMM2: [16 x 1024] @ [1024 x 128] ----------------------
    const int n0 = wave * 16;
    v8f acc2 = vzero;
    #pragma unroll 8
    for (int kk = 0; kk < 32; ++kk) {
        const int k0 = kk * 32;
        v16bf a2 = build_a(sM, lane, 0, k0, SMLD);              // sM is m-major
        v16bf b2 = build_b(WoutT, lane, n0, k0, 1024);          // WoutT is n-major
        acc2 = __builtin_amdgcn_wmma_f32_16x16x32_bf16(
            false, a2, false, b2, (short)0, acc2, false, false);
    }

    // ---------------- epilogue: + b_out, store f32 ---------------------------
    const int e  = n0 + (lane & 15);
    const float bo = b_out[e];
    #pragma unroll
    for (int r = 0; r < 8; ++r) {
        const int p  = (lane >> 4) * 8 + r;
        const int il = p >> 2;
        const int jl = p & 3;
        out[((i0 + il) * NRES + (j0 + jl)) * CZ + e] = acc2[r] + bo;
    }
}

// ---------------------------------------------------------------------------
// Launch
// ---------------------------------------------------------------------------
extern "C" void kernel_launch(void* const* d_in, const int* in_sizes, int n_in,
                              void* d_out, int out_size, void* d_ws, size_t ws_size,
                              hipStream_t stream) {
    const float* msa      = (const float*)d_in[0];
    const float* ln_gamma = (const float*)d_in[1];
    const float* ln_beta  = (const float*)d_in[2];
    const float* W_left   = (const float*)d_in[3];
    const float* b_left   = (const float*)d_in[4];
    const float* W_right  = (const float*)d_in[5];
    const float* b_right  = (const float*)d_in[6];
    const float* W_out    = (const float*)d_in[7];
    const float* b_out    = (const float*)d_in[8];
    float* out = (float*)d_out;

    unsigned short* Abf   = (unsigned short*)d_ws;            // 8192*128 bf16 = 2 MB
    unsigned short* Bbf   = Abf + (size_t)MTOT * S_DIM;       // 2 MB
    unsigned short* WoutT = Bbf + (size_t)MTOT * S_DIM;       // 128*1024 bf16 = 256 KB

    opm_ln_proj_kernel<<<S_DIM * NRES, 256, 0, stream>>>(
        msa, ln_gamma, ln_beta, W_left, b_left, W_right, b_right, Abf, Bbf);

    opm_wcvt_kernel<<<(1024 * CZ) / 256, 256, 0, stream>>>(W_out, WoutT);

    dim3 grid2(NRES / 4, NRES / 4);
    opm_fused_kernel<<<grid2, 256, 0, stream>>>(Abf, Bbf, WoutT, b_out, out);
}
